// GraphConvLayer_88974542504681
// MI455X (gfx1250) — matrix operand order repaired and checked by reading, not compiled, not run on hardware
//
#include <hip/hip_runtime.h>

#define N_NODES 50000
#define N_EDGES 600000
#define D       128
#define M_TILES (N_NODES / 16)   // 3125, exact

typedef __attribute__((ext_vector_type(16))) __bf16        v16bf;
typedef __attribute__((ext_vector_type(2)))  __bf16        v2bf;
typedef __attribute__((ext_vector_type(8)))  float         v8f;
typedef __attribute__((ext_vector_type(8)))  unsigned int  v8u;

// ---------- bf16 helpers: native casts -> v_cvt_pk_bf16_f32 ----------
__device__ __forceinline__ unsigned int pack2(float a, float b) {
  v2bf p;
  p.x = (__bf16)a;
  p.y = (__bf16)b;
  return __builtin_bit_cast(unsigned int, p);
}
__device__ __forceinline__ unsigned short bf16_bits(float f) {
  return __builtin_bit_cast(unsigned short, (__bf16)f);
}

// A-fragment for V_WMMA_F32_16X16X32_BF16, lane holds row m.
// half h: VGPR0-7 <- K = kk+16h .. +15
__device__ __forceinline__ v16bf load_a_frag(const float* __restrict__ row,
                                             int kk, int h, float s) {
  const float4* p = (const float4*)row;
  int c1 = (kk + 16 * h) >> 2;
  float4 f0 = p[c1], f1 = p[c1 + 1];
  float4 f2 = p[c1 + 2], f3 = p[c1 + 3];
  v8u u;
  u[0] = pack2(f0.x * s, f0.y * s); u[1] = pack2(f0.z * s, f0.w * s);
  u[2] = pack2(f1.x * s, f1.y * s); u[3] = pack2(f1.z * s, f1.w * s);
  u[4] = pack2(f2.x * s, f2.y * s); u[5] = pack2(f2.z * s, f2.w * s);
  u[6] = pack2(f3.x * s, f3.y * s); u[7] = pack2(f3.z * s, f3.w * s);
  return __builtin_bit_cast(v16bf, u);
}

// B-fragment: lane holds column n; half h holds K = kk+16h .. +15,
// 16 contiguous bf16 from the pre-transposed weight row.
__device__ __forceinline__ v16bf load_b_frag(const unsigned short* __restrict__ p) {
  uint4 a = ((const uint4*)p)[0];
  uint4 b = ((const uint4*)p)[1];
  v8u u;
  u[0] = a.x; u[1] = a.y; u[2] = a.z; u[3] = a.w;
  u[4] = b.x; u[5] = b.y; u[6] = b.z; u[7] = b.w;
  return __builtin_bit_cast(v16bf, u);
}

// ---------- kernels ----------

// W[k][n] (f32) -> Wt[n][k] (bf16), same for Wres.
__global__ __launch_bounds__(256) void convert_weights_kernel(
    const float* __restrict__ W, const float* __restrict__ Wres,
    unsigned short* __restrict__ Wt, unsigned short* __restrict__ Wrt) {
  int i = blockIdx.x * 256 + threadIdx.x;
  if (i < D * D) {
    int n = i >> 7, k = i & 127;
    Wt[i]  = bf16_bits(W[k * D + n]);
    Wrt[i] = bf16_bits(Wres[k * D + n]);
  }
}

__global__ __launch_bounds__(256) void degree_kernel(
    const int* __restrict__ src, const int* __restrict__ dst,
    float* __restrict__ deg_out, float* __restrict__ deg_in) {
  int i = blockIdx.x * 256 + threadIdx.x;
  if (i < N_EDGES) {
    atomicAdd(&deg_out[src[i]], 1.0f);
    atomicAdd(&deg_in[dst[i]], 1.0f);
  }
}

// One wave32 per edge: agg[dst] += X[src] * rsqrt(deg_out[src])
__global__ __launch_bounds__(256) void scatter_kernel(
    const float* __restrict__ X, const int* __restrict__ src,
    const int* __restrict__ dst, const float* __restrict__ deg_out,
    float* __restrict__ agg) {
  int lane = threadIdx.x & 31;
  int e = blockIdx.x * 8 + (threadIdx.x >> 5);
  if (e < N_EDGES) {
    int s = src[e], d = dst[e];
    float ns = rsqrtf(fmaxf(deg_out[s], 1.0f));
    float4 x = ((const float4*)(X + (size_t)s * D))[lane];
    float* ap = agg + (size_t)d * D + lane * 4;
    atomicAdd(ap + 0, x.x * ns);
    atomicAdd(ap + 1, x.y * ns);
    atomicAdd(ap + 2, x.z * ns);
    atomicAdd(ap + 3, x.w * ns);
  }
}

// Per wave: 16-row M-tile. gX = relu((agg*norm_dst)@W) + relu(X@Wres),
// store to out, accumulate per-column BN sums via LDS -> global atomics.
__global__ __launch_bounds__(256) void gemm_bn_kernel(
    const float* __restrict__ agg, const float* __restrict__ X,
    const float* __restrict__ deg_in,
    const unsigned short* __restrict__ Wt, const unsigned short* __restrict__ Wrt,
    float* __restrict__ out, float* __restrict__ gsum, float* __restrict__ gsq) {
  __shared__ float lsum[D];
  __shared__ float lsq[D];
  int tid = threadIdx.x;
  if (tid < D) { lsum[tid] = 0.0f; lsq[tid] = 0.0f; }
  __syncthreads();

  int wave = blockIdx.x * 8 + (tid >> 5);
  int lane = tid & 31;
  if (wave < M_TILES) {
    int m0 = wave * 16;
    int h = lane >> 4;
    int mrow = m0 + (lane & 15);
    float sdst = rsqrtf(fmaxf(deg_in[mrow], 1.0f));
    const float* arow = agg + (size_t)mrow * D;
    const float* xrow = X + (size_t)mrow * D;

    v16bf aA[4], aX[4];
#pragma unroll
    for (int k = 0; k < 4; ++k) {
      aA[k] = load_a_frag(arow, k * 32, h, sdst);
      aX[k] = load_a_frag(xrow, k * 32, h, 1.0f);
    }

    for (int nt = 0; nt < 8; ++nt) {
      int ncol = nt * 16 + (lane & 15);
      v8f acc1 = {};
      v8f acc2 = {};
#pragma unroll
      for (int k = 0; k < 4; ++k) {
        v16bf b1 = load_b_frag(Wt + (size_t)ncol * D + k * 32 + 16 * h);
        acc1 = __builtin_amdgcn_wmma_f32_16x16x32_bf16(
            false, aA[k], false, b1, (short)0, acc1, false, false);
        v16bf b2 = load_b_frag(Wrt + (size_t)ncol * D + k * 32 + 16 * h);
        acc2 = __builtin_amdgcn_wmma_f32_16x16x32_bf16(
            false, aX[k], false, b2, (short)0, acc2, false, false);
      }
      float s = 0.0f, q = 0.0f;
#pragma unroll
      for (int r = 0; r < 8; ++r) {
        float y = fmaxf(acc1[r], 0.0f) + fmaxf(acc2[r], 0.0f);
        out[(size_t)(m0 + r + 8 * h) * D + ncol] = y;
        s += y;
        q += y * y;
      }
      atomicAdd(&lsum[ncol], s);
      atomicAdd(&lsq[ncol], q);
    }
  }
  __syncthreads();
  if (tid < D) {
    atomicAdd(&gsum[tid], lsum[tid]);
    atomicAdd(&gsq[tid], lsq[tid]);
  }
}

__global__ void bn_finalize_kernel(
    const float* __restrict__ gsum, const float* __restrict__ gsq,
    const float* __restrict__ gamma, const float* __restrict__ beta,
    float* __restrict__ scale, float* __restrict__ shift) {
  int n = threadIdx.x;  // 128 threads
  const float invN = 1.0f / (float)N_NODES;
  float mean = gsum[n] * invN;
  float var = fmaxf(gsq[n] * invN - mean * mean, 0.0f);
  float inv = rsqrtf(var + 1e-5f);
  float sc = gamma[n] * inv;
  scale[n] = sc;
  shift[n] = beta[n] - mean * sc;
}

__global__ __launch_bounds__(256) void bn_apply_kernel(
    float* __restrict__ out, const float* __restrict__ scale,
    const float* __restrict__ shift) {
  int i = blockIdx.x * 256 + threadIdx.x;  // over N_NODES*32 float4
  if (i < N_NODES * 32) {
    int n4 = i & 31;
    float4 sc = ((const float4*)scale)[n4];
    float4 sh = ((const float4*)shift)[n4];
    float4 y = ((float4*)out)[i];
    y.x = y.x * sc.x + sh.x;
    y.y = y.y * sc.y + sh.y;
    y.z = y.z * sc.z + sh.z;
    y.w = y.w * sc.w + sh.w;
    ((float4*)out)[i] = y;
  }
}

extern "C" void kernel_launch(void* const* d_in, const int* in_sizes, int n_in,
                              void* d_out, int out_size, void* d_ws, size_t ws_size,
                              hipStream_t stream) {
  const float* X     = (const float*)d_in[0];
  const float* W     = (const float*)d_in[1];
  const float* Wres  = (const float*)d_in[2];
  const float* gamma = (const float*)d_in[3];
  const float* beta  = (const float*)d_in[4];
  const int*   src   = (const int*)d_in[5];
  const int*   dst   = (const int*)d_in[6];
  float* out = (float*)d_out;

  char* ws = (char*)d_ws;
  size_t off = 0;
  auto take = [&](size_t bytes) -> void* {
    void* p = ws + off;
    off += (bytes + 255) & ~(size_t)255;
    return p;
  };
  float* deg_out = (float*)take((size_t)N_NODES * 4);
  float* deg_in  = (float*)take((size_t)N_NODES * 4);
  float* agg     = (float*)take((size_t)N_NODES * D * 4);
  unsigned short* Wt  = (unsigned short*)take((size_t)D * D * 2);
  unsigned short* Wrt = (unsigned short*)take((size_t)D * D * 2);
  float* gsum  = (float*)take(D * 4);
  float* gsq   = (float*)take(D * 4);
  float* scale = (float*)take(D * 4);
  float* shift = (float*)take(D * 4);

  hipMemsetAsync(deg_out, 0, (size_t)N_NODES * 4, stream);
  hipMemsetAsync(deg_in, 0, (size_t)N_NODES * 4, stream);
  hipMemsetAsync(agg, 0, (size_t)N_NODES * D * 4, stream);
  hipMemsetAsync(gsum, 0, D * 4, stream);
  hipMemsetAsync(gsq, 0, D * 4, stream);

  convert_weights_kernel<<<(D * D + 255) / 256, 256, 0, stream>>>(W, Wres, Wt, Wrt);
  degree_kernel<<<(N_EDGES + 255) / 256, 256, 0, stream>>>(src, dst, deg_out, deg_in);
  scatter_kernel<<<(N_EDGES + 7) / 8, 256, 0, stream>>>(X, src, dst, deg_out, agg);
  gemm_bn_kernel<<<(M_TILES + 7) / 8, 256, 0, stream>>>(agg, X, deg_in, Wt, Wrt,
                                                        out, gsum, gsq);
  bn_finalize_kernel<<<1, D, 0, stream>>>(gsum, gsq, gamma, beta, scale, shift);
  bn_apply_kernel<<<(N_NODES * 32 + 255) / 256, 256, 0, stream>>>(out, scale, shift);
}
